// Handmade_conv2d_21998822490188
// MI455X (gfx1250) — matrix operand order repaired and checked by reading, not compiled, not run on hardware
//
#include <hip/hip_runtime.h>

typedef __attribute__((ext_vector_type(16))) __bf16        v16bf;
typedef __attribute__((ext_vector_type(8)))  float         v8f;
typedef __attribute__((ext_vector_type(8)))  unsigned int  v8u;

// Fixed shapes from the reference
#define C_IN      128
#define K_OUT     256
#define H_OUT     62
#define W_OUT     62
#define PQ        (H_OUT * W_OUT)        // 3844
#define SPATIAL   (32 * PQ)              // 123008 = 961 * 128
#define HW        4096                   // 64*64
#define X_N_STRIDE  (C_IN * HW)          // 524288
#define X_C_STRIDE  HW                   // 4096
#define W_OC_STRIDE (C_IN * 9)           // 1152
#define O_N_STRIDE  (K_OUT * PQ)         // 984064

#define XB_ELEMS  (32 * HW * C_IN)       // 16,777,216 bf16 (NHWC)
#define WB_ELEMS  (9 * K_OUT * C_IN)     // 294,912 bf16 ([tap][oc][c])
#define XB_BYTES  ((size_t)XB_ELEMS * 2) // 33,554,432
#define WB_BYTES  ((size_t)WB_ELEMS * 2) // 589,824
#define WS_NEEDED (XB_BYTES + WB_BYTES)

// Async global->LDS path (ASYNCcnt-tracked), guarded so compile never regresses
#if defined(__has_builtin)
#  if __has_builtin(__builtin_amdgcn_global_load_async_to_lds_b64) && \
      __has_builtin(__builtin_amdgcn_s_wait_asynccnt)
#    define HAVE_ASYNC 1
#  endif
#endif
#ifndef HAVE_ASYNC
#  define HAVE_ASYNC 0
#endif

// From the round-3 diagnostic, the builtin's parameters are:
//   (int __vector(2) addrspace(1)*, int __vector(2) addrspace(3)*, imm, imm)
typedef int v2i_t __attribute__((vector_size(8)));
typedef __attribute__((address_space(1))) v2i_t* as1_v2i;
typedef __attribute__((address_space(3))) v2i_t* as3_v2i;

__device__ __forceinline__ void stage_copy8(const unsigned short* g,
                                            unsigned short* l) {
#if HAVE_ASYNC
  // Flat LDS address: low 32 bits are the LDS offset (aperture layout, ISA 10.2)
  __builtin_amdgcn_global_load_async_to_lds_b64(
      (as1_v2i)(unsigned long long)g,
      (as3_v2i)(unsigned int)(unsigned long long)l, 0, 0);
#else
  *(uint2*)l = *(const uint2*)g;
#endif
}

__device__ __forceinline__ unsigned short f2bf(float f) {
  unsigned int u = __builtin_bit_cast(unsigned int, f);
  u += 0x7FFFu + ((u >> 16) & 1u);       // round-to-nearest-even
  return (unsigned short)(u >> 16);
}

// ---------------------------------------------------------------------------
// Prep kernel 1: X fp32 NCHW -> bf16 NHWC (LDS-tiled 32x32 transpose)
// ---------------------------------------------------------------------------
__global__ __launch_bounds__(256) void x_to_bf16_nhwc(
    const float* __restrict__ X, unsigned short* __restrict__ Xb) {
  __shared__ unsigned short t[32][33];
  const int lx = threadIdx.x & 31;
  const int ly = threadIdx.x >> 5;       // 0..7
  const int hw0 = blockIdx.x * 32;       // 128 tiles over hw=4096
  const int c0  = blockIdx.y * 32;       // 4 tiles over c=128
  const int n   = blockIdx.z;            // 32 images
#pragma unroll
  for (int j = 0; j < 32; j += 8)        // read: lanes walk hw (coalesced)
    t[ly + j][lx] = f2bf(X[(size_t)n * X_N_STRIDE +
                           (size_t)(c0 + ly + j) * X_C_STRIDE + hw0 + lx]);
  __syncthreads();
#pragma unroll
  for (int j = 0; j < 32; j += 8)        // write: lanes walk c (coalesced)
    Xb[((size_t)n * HW + hw0 + ly + j) * C_IN + c0 + lx] = t[lx][ly + j];
}

// ---------------------------------------------------------------------------
// Prep kernel 2: W fp32 [oc][c][r][s] -> bf16 [tap][oc][c]
// ---------------------------------------------------------------------------
__global__ __launch_bounds__(256) void w_to_bf16_tapocc(
    const float* __restrict__ W, unsigned short* __restrict__ Wb) {
  const int idx = blockIdx.x * 256 + threadIdx.x;  // < 294912
  const int tap = idx >> 15;
  const int rem = idx & 32767;
  const int oc  = rem >> 7;
  const int c   = rem & 127;
  Wb[idx] = f2bf(W[(size_t)oc * W_OC_STRIDE + c * 9 + tap]);
}

// ---------------------------------------------------------------------------
// Main kernel: implicit-GEMM conv, per-tap K loop, bf16 WMMA f32 accumulate.
// Double-buffered LDS; async global->LDS staging when available.
// One barrier per K-stage. LDS pitch 72B (b64-aligned, bank-conflict free).
// ---------------------------------------------------------------------------
#define RS 36   // LDS row stride in halves
#define NSTAGE 36

__global__ __launch_bounds__(256) void conv3x3_wmma_pre(
    const unsigned short* __restrict__ Xb,   // [n][h][w][c] bf16
    const unsigned short* __restrict__ Wb,   // [tap][oc][c] bf16
    float* __restrict__ Out) {
  __shared__ __align__(16) unsigned short aTb[2][128 * RS];  // weights [oc][c]
  __shared__ __align__(16) unsigned short bTb[2][128 * RS];  // input   [sp][c]

  const int tid    = threadIdx.x;
  const int lane   = tid & 31;
  const int wv     = tid >> 5;      // wave 0..7
  const int wm     = wv >> 2;       // 0..1 : 64 oc each
  const int wn     = wv & 3;        // 0..3 : 32 sp each
  const int mBase  = blockIdx.y * 128;
  const int spBase = blockIdx.x * 128;

  // Per-thread staging: 4 chunks of 8B (4 bf16) each for A and for B tile.
  unsigned aOffE[4], bOffE[4], stI[4];
#pragma unroll
  for (int i = 0; i < 4; ++i) {
    const int ch   = tid + i * 256;          // 0..1023
    const int row  = ch >> 3;                // 0..127
    const int part = ch & 7;                 // 8B chunk within 64B row
    stI[i]   = (unsigned)(row * RS + part * 4);
    aOffE[i] = (unsigned)(mBase + row) * 128u + (unsigned)part * 4u;
    const int sp  = spBase + row;
    const int n   = sp / PQ;
    const int rem = sp - n * PQ;
    const int p   = rem / W_OUT;
    const int q   = rem - p * W_OUT;
    bOffE[i] = ((unsigned)n * HW + (unsigned)p * 64u + (unsigned)q) * 128u +
               (unsigned)part * 4u;
  }

  v8f acc[4][2];
#pragma unroll
  for (int mt = 0; mt < 4; ++mt)
#pragma unroll
    for (int nt = 0; nt < 2; ++nt)
      acc[mt][nt] = (v8f){0.f, 0.f, 0.f, 0.f, 0.f, 0.f, 0.f, 0.f};

  // Prologue: stage 0 into buffer 0 (tap=0, kc=0 -> aAdd=0, bAdd=0)
#pragma unroll
  for (int i = 0; i < 4; ++i) stage_copy8(Wb + aOffE[i], aTb[0] + stI[i]);
#pragma unroll
  for (int i = 0; i < 4; ++i) stage_copy8(Xb + bOffE[i], bTb[0] + stI[i]);

  for (int t = 0; t < NSTAGE; ++t) {
#if HAVE_ASYNC
    __builtin_amdgcn_s_wait_asynccnt(0);   // this wave's copies landed in LDS
#endif
    __syncthreads();                        // all waves' copies visible

    // Issue stage t+1 into the other buffer (safe: past the barrier, so no
    // wave can still be reading buffer (t+1)&1 from stage t-1).
    if (t + 1 < NSTAGE) {
      const int tn  = t + 1;
      const int tap = tn >> 2;
      const int kc  = tn & 3;
      const int r   = tap / 3;
      const int s   = tap - r * 3;
      const unsigned aAdd = (unsigned)tap * 32768u + (unsigned)kc * 32u;
      const unsigned bAdd = (unsigned)(r * 64 + s) * 128u + (unsigned)kc * 32u;
      unsigned short* aN = aTb[tn & 1];
      unsigned short* bN = bTb[tn & 1];
#pragma unroll
      for (int i = 0; i < 4; ++i)
        stage_copy8(Wb + aOffE[i] + aAdd, aN + stI[i]);
#pragma unroll
      for (int i = 0; i < 4; ++i)
        stage_copy8(Xb + bOffE[i] + bAdd, bN + stI[i]);
      __builtin_prefetch(Xb + bOffE[0] + bAdd + 32u, 0, 1);
      __builtin_prefetch(Wb + aOffE[0] + aAdd + 32u, 0, 1);
    }

    const unsigned short* aT = aTb[t & 1];
    const unsigned short* bT = bTb[t & 1];

    // Fragment loads per documented gfx1250 WMMA VGPR layouts
    v16bf afr[4], bfr[2];
#pragma unroll
    for (int mt = 0; mt < 4; ++mt) {
      const int row = wm * 64 + mt * 16 + (lane & 15);
      v8u u;
#pragma unroll
      for (int v = 0; v < 8; ++v) {
        const int k = ((v >> 2) << 4) + ((lane >> 4) << 3) + ((v & 3) << 1);
        u[v] = *(const unsigned int*)(aT + row * RS + k);
      }
      afr[mt] = __builtin_bit_cast(v16bf, u);
    }
#pragma unroll
    for (int nt = 0; nt < 2; ++nt) {
      const int col = wn * 32 + nt * 16 + (lane & 15);
      v8u u;
#pragma unroll
      for (int v = 0; v < 8; ++v) {
        const int k = ((v >> 2) << 4) + ((lane >> 4) << 3) + ((v & 3) << 1);
        u[v] = *(const unsigned int*)(bT + col * RS + k);
      }
      bfr[nt] = __builtin_bit_cast(v16bf, u);
    }

#pragma unroll
    for (int mt = 0; mt < 4; ++mt)
#pragma unroll
      for (int nt = 0; nt < 2; ++nt)
        acc[mt][nt] = __builtin_amdgcn_wmma_f32_16x16x32_bf16(
            false, afr[mt], false, bfr[nt], (short)0, acc[mt][nt],
            false, false);
  }

  // Writeback: lane holds N=lane%16, VGPR j holds M=j+8*(lane/16)
#pragma unroll
  for (int nt = 0; nt < 2; ++nt) {
    const int sp  = spBase + wn * 32 + nt * 16 + (lane & 15);
    const int n   = sp / PQ;
    const int rem = sp - n * PQ;
    const unsigned outBase = (unsigned)n * O_N_STRIDE + (unsigned)rem;
#pragma unroll
    for (int mt = 0; mt < 4; ++mt) {
      const int m0 = mBase + wm * 64 + mt * 16 + ((lane >> 4) << 3);
#pragma unroll
      for (int j = 0; j < 8; ++j)
        Out[outBase + (unsigned)(m0 + j) * (unsigned)PQ] = acc[mt][nt][j];
    }
  }
}

// ---------------------------------------------------------------------------
// Fallback (no workspace): fused kernel, fp32->bf16 inline
// ---------------------------------------------------------------------------
#define ROW_STRIDE 34

__global__ __launch_bounds__(256) void conv3x3_wmma_fused(
    const float* __restrict__ X, const float* __restrict__ W,
    float* __restrict__ Out) {
  __shared__ unsigned short aT[128 * ROW_STRIDE];
  __shared__ unsigned short bT[128 * ROW_STRIDE];

  const int tid    = threadIdx.x;
  const int lane   = tid & 31;
  const int wv     = tid >> 5;
  const int wm     = wv >> 2;
  const int wn     = wv & 3;
  const int mBase  = blockIdx.y * 128;
  const int spBase = blockIdx.x * 128;

  unsigned int   aOff[16], bOff[16];
  unsigned short aSt[16],  bSt[16];
#pragma unroll
  for (int i = 0; i < 16; ++i) {
    const int idx = tid + i * 256;
    const int ocl = idx >> 5, cl = idx & 31;
    aOff[i] = (unsigned)(mBase + ocl) * W_OC_STRIDE + (unsigned)cl * 9u;
    aSt[i]  = (unsigned short)(ocl * ROW_STRIDE + cl);
    const int cl2 = idx >> 7, spl = idx & 127;
    const int sp  = spBase + spl;
    const int n   = sp / PQ;
    const int rem = sp - n * PQ;
    const int p   = rem / W_OUT;
    const int q   = rem - p * W_OUT;
    bOff[i] = (unsigned)n * X_N_STRIDE + (unsigned)cl2 * X_C_STRIDE +
              (unsigned)p * 64u + (unsigned)q;
    bSt[i]  = (unsigned short)(spl * ROW_STRIDE + cl2);
  }

  v8f acc[4][2];
#pragma unroll
  for (int mt = 0; mt < 4; ++mt)
#pragma unroll
    for (int nt = 0; nt < 2; ++nt)
      acc[mt][nt] = (v8f){0.f, 0.f, 0.f, 0.f, 0.f, 0.f, 0.f, 0.f};

  for (int tap = 0; tap < 9; ++tap) {
    const int r = tap / 3;
    const int s = tap - r * 3;
    const unsigned rsOff = (unsigned)(r * 64 + s);
    for (int kc = 0; kc < 4; ++kc) {
      const unsigned c0 = (unsigned)kc * 32u;
#pragma unroll
      for (int i = 0; i < 16; ++i)
        aT[aSt[i]] = f2bf(W[aOff[i] + c0 * 9u + (unsigned)tap]);
#pragma unroll
      for (int i = 0; i < 16; ++i)
        bT[bSt[i]] = f2bf(X[bOff[i] + c0 * X_C_STRIDE + rsOff]);
      __syncthreads();

      v16bf afr[4], bfr[2];
#pragma unroll
      for (int mt = 0; mt < 4; ++mt) {
        const int row = wm * 64 + mt * 16 + (lane & 15);
        v8u u;
#pragma unroll
        for (int v = 0; v < 8; ++v) {
          const int k = ((v >> 2) << 4) + ((lane >> 4) << 3) + ((v & 3) << 1);
          u[v] = *(const unsigned int*)(aT + row * ROW_STRIDE + k);
        }
        afr[mt] = __builtin_bit_cast(v16bf, u);
      }
#pragma unroll
      for (int nt = 0; nt < 2; ++nt) {
        const int col = wn * 32 + nt * 16 + (lane & 15);
        v8u u;
#pragma unroll
        for (int v = 0; v < 8; ++v) {
          const int k = ((v >> 2) << 4) + ((lane >> 4) << 3) + ((v & 3) << 1);
          u[v] = *(const unsigned int*)(bT + col * ROW_STRIDE + k);
        }
        bfr[nt] = __builtin_bit_cast(v16bf, u);
      }
#pragma unroll
      for (int mt = 0; mt < 4; ++mt)
#pragma unroll
        for (int nt = 0; nt < 2; ++nt)
          acc[mt][nt] = __builtin_amdgcn_wmma_f32_16x16x32_bf16(
              false, afr[mt], false, bfr[nt], (short)0, acc[mt][nt],
              false, false);
      __syncthreads();
    }
  }

#pragma unroll
  for (int nt = 0; nt < 2; ++nt) {
    const int sp  = spBase + wn * 32 + nt * 16 + (lane & 15);
    const int n   = sp / PQ;
    const int rem = sp - n * PQ;
    const unsigned outBase = (unsigned)n * O_N_STRIDE + (unsigned)rem;
#pragma unroll
    for (int mt = 0; mt < 4; ++mt) {
      const int m0 = mBase + wm * 64 + mt * 16 + ((lane >> 4) << 3);
#pragma unroll
      for (int j = 0; j < 8; ++j)
        Out[outBase + (unsigned)(m0 + j) * (unsigned)PQ] = acc[mt][nt][j];
    }
  }
}

extern "C" void kernel_launch(void* const* d_in, const int* in_sizes, int n_in,
                              void* d_out, int out_size, void* d_ws, size_t ws_size,
                              hipStream_t stream) {
  const float* X = (const float*)d_in[0];   // (32,128,64,64) fp32
  const float* W = (const float*)d_in[1];   // (256,128,3,3)  fp32
  float* Out = (float*)d_out;               // (32,256,62,62) fp32

  if (d_ws != nullptr && ws_size >= WS_NEEDED) {
    unsigned short* Xb = (unsigned short*)d_ws;
    unsigned short* Wb = (unsigned short*)((char*)d_ws + XB_BYTES);
    x_to_bf16_nhwc<<<dim3(HW / 32, C_IN / 32, 32), 256, 0, stream>>>(X, Xb);
    w_to_bf16_tapocc<<<WB_ELEMS / 256, 256, 0, stream>>>(W, Wb);
    conv3x3_wmma_pre<<<dim3(SPATIAL / 128, K_OUT / 128), 256, 0, stream>>>(
        Xb, Wb, Out);
  } else {
    conv3x3_wmma_fused<<<dim3(SPATIAL / 128, K_OUT / 128), 256, 0, stream>>>(
        X, W, Out);
  }
}
